// ProfessionalHierarchicalUNet_40123584479845
// MI455X (gfx1250) — matrix operand order repaired and checked by reading, not compiled, not run on hardware
//
#include <hip/hip_runtime.h>
#include <math.h>

// ---------------------------------------------------------------------------
// CDNA5 (gfx1250) types for WMMA
// ---------------------------------------------------------------------------
typedef __attribute__((ext_vector_type(16))) __bf16 v16bf;
typedef __attribute__((ext_vector_type(8)))  __bf16 v8bf;
typedef __attribute__((ext_vector_type(8)))  float  v8f;

__device__ __forceinline__ float gelu_f(float x) {
    // tanh approximation (jax.nn.gelu default approximate=True)
    float x3 = x * x * x;
    return 0.5f * x * (1.0f + tanhf(0.7978845608028654f * (x + 0.044715f * x3)));
}

__device__ __forceinline__ unsigned pack_bf16x2(float a, float b) {
    union { __bf16 h[2]; unsigned u; } p;
    p.h[0] = (__bf16)a;   // lowers to v_cvt_pk_bf16_f32
    p.h[1] = (__bf16)b;
    return p.u;
}

// ---------------------------------------------------------------------------
// Batched GEMM: C[bz] = act(alpha * A[bz] @ B[bz] + bias)
//   A: [M,K] f32 (row stride lda, batch stride sA)
//   B: [K,N] f32 (row stride ldb) or [N,K] if transB
//   C: [M,N] f32
// Block: 128 threads = 4 waves. Macro tile 32(M) x 64(N), K step 64.
// Waves 2x2: each wave owns 16x32 -> 2 accumulators, 4 WMMA / K-iter.
// LDS: A as [m][k], B as [n][k] (transposed) so BOTH fragment gathers are
// contiguous 16-byte ds_load_b128 pairs (per ISA 7.12.2 layouts: a lane's
// A-frag = two 8-elem runs of one m-row; B-frag = two 8-elem runs of one n-row).
// Rows padded to 72 bf16 (144B) for LDS bank stagger, 16B alignment kept.
// act: 0 = none, 1 = gelu, 2 = 120*sigmoid
// ---------------------------------------------------------------------------
#define LDS_STRIDE 72

__global__ __launch_bounds__(128) void wmma_gemm_k(
    int M, int N, int K,
    const float* __restrict__ A, long long lda, long long sA,
    const float* __restrict__ B, long long ldb, long long sB, int transB,
    float* __restrict__ C, long long ldc, long long sC,
    float alpha, const float* __restrict__ bias, int act)
{
    __shared__ __bf16 lA [32 * LDS_STRIDE];   // [m][k]
    __shared__ __bf16 lBt[64 * LDS_STRIDE];   // [n][k]

    const int tid  = threadIdx.x;
    const int lane = tid & 31;
    const int wv   = tid >> 5;            // wave id 0..3
    const int wm   = (wv >> 1) << 4;      // wave M offset (0/16)
    const int wn   = (wv & 1) << 5;       // wave N offset (0/32)

    const int n0 = blockIdx.x * 64;
    const int m0 = blockIdx.y * 32;
    const long long bz = blockIdx.z;

    const float* Ab = A + bz * sA;
    const float* Bb = B + bz * sB;
    float*       Cb = C + bz * sC;

    v8f acc0 = {};
    v8f acc1 = {};

    const int col  = lane & 15;
    const int half = lane >> 4;           // 0: lanes 0-15, 1: lanes 16-31
    const int koffA = half << 3;          // A frag: +8 for upper lanes
    const int koffB = half << 3;          // B frag: +8 for upper lanes

    for (int k0 = 0; k0 < K; k0 += 64) {
        // ---------------- stage A tile: 32 x 64 -> lA[m][k] ----------------
        if (m0 + 32 <= M && k0 + 64 <= K) {
#pragma unroll
            for (int it = 0; it < 8; ++it) {            // 1024 pairs / 128 thr
                int idx = tid + it * 128;
                int r = idx >> 5, c2 = (idx & 31) << 1;
                const float* s = Ab + (long long)(m0 + r) * lda + k0 + c2;
                *(unsigned*)&lA[r * LDS_STRIDE + c2] = pack_bf16x2(s[0], s[1]);
            }
        } else {
            for (int idx = tid; idx < 2048; idx += 128) {
                int r = idx >> 6, c = idx & 63;
                int mm = m0 + r, kk = k0 + c;
                float v = (mm < M && kk < K) ? Ab[(long long)mm * lda + kk] : 0.0f;
                lA[r * LDS_STRIDE + c] = (__bf16)v;
            }
        }
        // ---------------- stage B tile: 64 x 64 -> lBt[n][k] ----------------
        if (k0 + 64 <= K && n0 + 64 <= N) {
            if (transB) {
                // global contiguous along k, LDS contiguous along k: ideal
#pragma unroll
                for (int it = 0; it < 16; ++it) {       // 2048 pairs / 128 thr
                    int idx = tid + it * 128;
                    int n = idx >> 5, kp = (idx & 31) << 1;
                    const float* s = Bb + (long long)(n0 + n) * ldb + k0 + kp;
                    *(unsigned*)&lBt[n * LDS_STRIDE + kp] = pack_bf16x2(s[0], s[1]);
                }
            } else {
                // two strided global loads, one packed LDS store
#pragma unroll
                for (int it = 0; it < 16; ++it) {
                    int idx = tid + it * 128;
                    int n = idx >> 5, kp = (idx & 31) << 1;
                    const float* s = Bb + (long long)(k0 + kp) * ldb + n0 + n;
                    *(unsigned*)&lBt[n * LDS_STRIDE + kp] = pack_bf16x2(s[0], s[ldb]);
                }
            }
        } else {
            for (int idx = tid; idx < 4096; idx += 128) {
                int n = idx >> 6, k = idx & 63;
                int kk = k0 + k, nn = n0 + n;
                float v = 0.0f;
                if (kk < K && nn < N)
                    v = transB ? Bb[(long long)nn * ldb + kk]
                               : Bb[(long long)kk * ldb + nn];
                lBt[n * LDS_STRIDE + k] = (__bf16)v;
            }
        }
        __syncthreads();

        // -------- 2 K-subtiles x 2 N-subtiles = 4 WMMA, b128 gathers --------
#pragma unroll
        for (int ks = 0; ks < 64; ks += 32) {
            const __bf16* arow = &lA[(wm + col) * LDS_STRIDE + ks];
            v8bf alo = *(const v8bf*)(arow + koffA);
            v8bf ahi = *(const v8bf*)(arow + 16 + koffA);
            v16bf af = __builtin_shufflevector(alo, ahi,
                0, 1, 2, 3, 4, 5, 6, 7, 8, 9, 10, 11, 12, 13, 14, 15);

            const __bf16* b0p = &lBt[(wn + col) * LDS_STRIDE + ks + koffB];
            v8bf b0lo = *(const v8bf*)(b0p);
            v8bf b0hi = *(const v8bf*)(b0p + 16);
            v16bf b0 = __builtin_shufflevector(b0lo, b0hi,
                0, 1, 2, 3, 4, 5, 6, 7, 8, 9, 10, 11, 12, 13, 14, 15);

            const __bf16* b1p = &lBt[(wn + 16 + col) * LDS_STRIDE + ks + koffB];
            v8bf b1lo = *(const v8bf*)(b1p);
            v8bf b1hi = *(const v8bf*)(b1p + 16);
            v16bf b1 = __builtin_shufflevector(b1lo, b1hi,
                0, 1, 2, 3, 4, 5, 6, 7, 8, 9, 10, 11, 12, 13, 14, 15);

            acc0 = __builtin_amdgcn_wmma_f32_16x16x32_bf16(
                false, af, false, b0, (short)0, acc0, false, false);
            acc1 = __builtin_amdgcn_wmma_f32_16x16x32_bf16(
                false, af, false, b1, (short)0, acc1, false, false);
        }
        __syncthreads();
    }

    // ---- epilogue + store (C/D layout: VGPR r -> M=r(+8), N=lane&15) ----
    const int mbase = m0 + wm + (half << 3);
#pragma unroll
    for (int sub = 0; sub < 2; ++sub) {
        const v8f& acc = sub ? acc1 : acc0;
        int n = n0 + wn + sub * 16 + col;
#pragma unroll
        for (int r = 0; r < 8; ++r) {
            int m = mbase + r;
            if (m < M && n < N) {
                float v = acc[r] * alpha;
                if (bias) v += bias[n];
                if (act == 1)      v = gelu_f(v);
                else if (act == 2) v = 120.0f / (1.0f + expf(-v));
                Cb[(long long)m * ldc + n] = v;
            }
        }
    }
}

// ---------------------------------------------------------------------------
// Direct conv1d: y[b,o,l] = act(bias[o] + sum_{i,k} w[o,i,k] x[b,i,l*s+k-pad])
// Whole model fits in 192MB L2, so naive global reads hit L2.
// ---------------------------------------------------------------------------
__global__ void conv1d_k(const float* __restrict__ x, const float* __restrict__ w,
                         const float* __restrict__ b, float* __restrict__ y,
                         int Bn, int Cin, int Cout, int Lin, int Lout,
                         int Kw, int strd, int pad, int act)
{
    long long i = (long long)blockIdx.x * blockDim.x + threadIdx.x;
    long long tot = (long long)Bn * Cout * Lout;
    if (i >= tot) return;
    int l = (int)(i % Lout);
    long long t = i / Lout;
    int o  = (int)(t % Cout);
    int bb = (int)(t / Cout);
    const float* xb = x + (long long)bb * Cin * Lin;
    const float* wo = w + (long long)o * Cin * Kw;
    float acc = b[o];
    int base = l * strd - pad;
    for (int ci = 0; ci < Cin; ++ci) {
        const float* xc = xb + (long long)ci * Lin;
        const float* wc = wo + ci * Kw;
        for (int kk = 0; kk < Kw; ++kk) {
            int p = base + kk;
            if (p >= 0 && p < Lin) acc += wc[kk] * xc[p];
        }
    }
    if (act) acc = gelu_f(acc);
    y[i] = acc;
}

// S4 kernel synthesis: K[c,l] = sum_n B[c,n]*C[c,n]*exp(-softplus(A[c,n])*l)
__global__ void ssm_kfun_k(const float* __restrict__ lA, const float* __restrict__ Bm,
                           const float* __restrict__ Cm, float* __restrict__ Kout,
                           int C, int N, int L)
{
    long long i = (long long)blockIdx.x * blockDim.x + threadIdx.x;
    if (i >= (long long)C * L) return;
    int l = (int)(i % L);
    int c = (int)(i / L);
    float t = (float)l;
    float acc = 0.0f;
    for (int n = 0; n < N; ++n) {
        float a = lA[c * N + n];
        float sp = (a > 20.0f) ? a : log1pf(expf(a));
        acc += Bm[c * N + n] * Cm[c * N + n] * expf(-sp * t);
    }
    Kout[i] = acc;
}

// Causal conv + fused residual: y = x*(1+D[c]) + sum_{t<=l} K[c,l-t]*x[b,c,t]
__global__ __launch_bounds__(256) void ssm_apply_k(
    const float* __restrict__ x, const float* __restrict__ Kc,
    const float* __restrict__ D, float* __restrict__ y, int C, int L)
{
    __shared__ float xs[256];
    int tid = threadIdx.x;
    int l = blockIdx.x * 256 + tid;
    int c = blockIdx.y, bb = blockIdx.z;
    const float* xr = x + ((long long)bb * C + c) * L;
    const float* Kr = Kc + (long long)c * L;
    float acc = 0.0f;
    int lmax = blockIdx.x * 256 + 255;
    if (lmax >= L) lmax = L - 1;
    for (int t0 = 0; t0 <= lmax; t0 += 256) {
        int tt = t0 + tid;
        xs[tid] = (tt < L) ? xr[tt] : 0.0f;
        __syncthreads();
        int te = lmax - t0; if (te > 255) te = 255;
        for (int j = 0; j <= te; ++j) {
            int t = t0 + j;
            if (l < L && t <= l) acc += xs[j] * Kr[l - t];
        }
        __syncthreads();
    }
    if (l < L) y[((long long)bb * C + c) * L + l] = acc + xr[l] * (1.0f + D[c]);
}

// dir 0: [B,C,L] -> [B,L,C]   dir 1: [B,L,C] -> [B,C,L]
__global__ void transpose_k(const float* __restrict__ x, float* __restrict__ y,
                            int Bn, int C, int L, int dir)
{
    long long i = (long long)blockIdx.x * blockDim.x + threadIdx.x;
    long long tot = (long long)Bn * C * L;
    if (i >= tot) return;
    int l = (int)(i % L);
    long long t = i / L;
    int c = (int)(t % C);
    int b = (int)(t / C);
    long long cl = ((long long)b * C + c) * L + l;
    long long lc = ((long long)b * L + l) * C + c;
    if (dir == 0) y[lc] = x[cl];
    else          y[cl] = x[lc];
}

__device__ __forceinline__ float block_sum(float v, float* red) {
    int t = threadIdx.x;
    red[t] = v; __syncthreads();
    for (int off = 128; off > 0; off >>= 1) {
        if (t < off) red[t] += red[t + off];
        __syncthreads();
    }
    float r = red[0]; __syncthreads();
    return r;
}

// y = g * (x+res - mu)/sqrt(var+eps) + b, one block (256 thr) per row; in-place ok
__global__ __launch_bounds__(256) void layernorm_k(
    const float* __restrict__ x, const float* __restrict__ res,
    const float* __restrict__ g, const float* __restrict__ b,
    float* __restrict__ y, int D)
{
    __shared__ float red[256];
    long long row = blockIdx.x;
    const float* xr = x + row * D;
    const float* rr = res ? res + row * D : nullptr;
    float s = 0.0f;
    for (int j = threadIdx.x; j < D; j += 256) s += xr[j] + (rr ? rr[j] : 0.0f);
    float mu = block_sum(s, red) / D;
    float s2 = 0.0f;
    for (int j = threadIdx.x; j < D; j += 256) {
        float v = xr[j] + (rr ? rr[j] : 0.0f) - mu;
        s2 += v * v;
    }
    float inv = rsqrtf(block_sum(s2, red) / D + 1e-5f);
    for (int j = threadIdx.x; j < D; j += 256) {
        float v = xr[j] + (rr ? rr[j] : 0.0f);
        y[row * D + j] = g[j] * (v - mu) * inv + b[j];
    }
}

// in-place row softmax, one block per row
__global__ __launch_bounds__(256) void softmax_rows_k(float* __restrict__ x, int Dn)
{
    __shared__ float red[256];
    long long row = blockIdx.x;
    float* xr = x + row * Dn;
    int t = threadIdx.x;
    float m = -1e30f;
    for (int j = t; j < Dn; j += 256) m = fmaxf(m, xr[j]);
    red[t] = m; __syncthreads();
    for (int off = 128; off > 0; off >>= 1) {
        if (t < off) red[t] = fmaxf(red[t], red[t + off]);
        __syncthreads();
    }
    float mx = red[0]; __syncthreads();
    float s = 0.0f;
    for (int j = t; j < Dn; j += 256) s += expf(xr[j] - mx);
    float inv = 1.0f / block_sum(s, red);
    for (int j = t; j < Dn; j += 256) xr[j] = expf(xr[j] - mx) * inv;
}

// FiLM apply: x = x*(1+g[b,c]) + be[b,c], gb = [B, 2C]
__global__ void film_k(float* __restrict__ x, const float* __restrict__ gb,
                       int C, int L, int Bn)
{
    long long i = (long long)blockIdx.x * blockDim.x + threadIdx.x;
    long long tot = (long long)Bn * C * L;
    if (i >= tot) return;
    long long t = i / L;
    int c = (int)(t % C);
    int b = (int)(t / C);
    float g  = gb[(long long)b * 2 * C + c];
    float be = gb[(long long)b * 2 * C + C + c];
    x[i] = x[i] * (1.0f + g) + be;
}

__global__ void upsample2_k(const float* __restrict__ x, float* __restrict__ y,
                            int Bn, int C, int L)
{
    long long i = (long long)blockIdx.x * blockDim.x + threadIdx.x;
    long long tot = (long long)Bn * C * 2 * L;
    if (i >= tot) return;
    int l2 = (int)(i % (2 * L));
    long long t = i / (2 * L);
    y[i] = x[t * L + (l2 >> 1)];
}

__global__ void concat_k(const float* __restrict__ a, const float* __restrict__ s,
                         float* __restrict__ y, int Bn, int Ca, int Cb, int L)
{
    long long i = (long long)blockIdx.x * blockDim.x + threadIdx.x;
    int Ct = Ca + Cb;
    long long tot = (long long)Bn * Ct * L;
    if (i >= tot) return;
    int l = (int)(i % L);
    long long t = i / L;
    int c = (int)(t % Ct);
    int b = (int)(t / Ct);
    y[i] = (c < Ca) ? a[((long long)b * Ca + c) * L + l]
                    : s[((long long)b * Cb + (c - Ca)) * L + l];
}

// logits[b,l] = sum_c x[b,c,l] * w[c]
__global__ void pool_logits_k(const float* __restrict__ x, const float* __restrict__ w,
                              float* __restrict__ out, int Bn, int C, int L)
{
    long long i = (long long)blockIdx.x * blockDim.x + threadIdx.x;
    if (i >= (long long)Bn * L) return;
    int l = (int)(i % L);
    int b = (int)(i / L);
    float acc = 0.0f;
    for (int c = 0; c < C; ++c) acc += x[((long long)b * C + c) * L + l] * w[c];
    out[i] = acc;
}

// out[b,c] = sum_l x[b,c,l] * s[b,l]
__global__ void pool_reduce_k(const float* __restrict__ x, const float* __restrict__ s,
                              float* __restrict__ out, int Bn, int C, int L)
{
    long long i = (long long)blockIdx.x * blockDim.x + threadIdx.x;
    if (i >= (long long)Bn * C) return;
    int c = (int)(i % C);
    int b = (int)(i / C);
    const float* xr = x + ((long long)b * C + c) * L;
    const float* sr = s + (long long)b * L;
    float acc = 0.0f;
    for (int l = 0; l < L; ++l) acc += xr[l] * sr[l];
    out[i] = acc;
}

// ---------------------------------------------------------------------------
// Host orchestration
// ---------------------------------------------------------------------------
extern "C" void kernel_launch(void* const* d_in, const int* in_sizes, int n_in,
                              void* d_out, int out_size, void* d_ws, size_t ws_size,
                              hipStream_t stream)
{
    (void)in_sizes; (void)n_in; (void)out_size; (void)ws_size;
    const float* X  = (const float*)d_in[0];   // [8,1,2048]
    const float* SP = (const float*)d_in[1];   // [8,4]
    auto in = [&](int i) { return (const float*)d_in[i]; };
    float* OUT = (float*)d_out;
    float* ws  = (float*)d_ws;

    // ---- workspace carve-out (floats) ----
    size_t off = 0;
    auto alloc = [&](size_t n) { float* p = ws + off; off += (n + 63) & ~(size_t)63; return p; };
    float* bSE  = alloc(64);            // gelu(static @ se_w + se_b)  [8,8]
    float* bGB  = alloc(8192);          // FiLM gamma/beta             [8,2C]
    float* P0   = alloc(2097152);       // ping-pong activation [B,C,L]
    float* P1   = alloc(2097152);
    float* P2   = alloc(2097152);
    float* SK[4]; for (int i = 0; i < 4; ++i) SK[i] = alloc(1048576);
    float* bK   = alloc(131072);        // S4 kernel K[c,l]
    float* bT   = alloc(1048576);       // transposed activations [B,L,C]
    float* bTMP = alloc(1048576);       // attention output
    float* bFF  = alloc(2097152);       // FFN hidden
    float* bQ   = alloc(1048576);
    float* bKK  = alloc(1048576);
    float* bV   = alloc(1048576);
    float* bSC  = alloc(8388608);       // per-batch scores [H,L,L] (<=32MB)
    float* bSIG = alloc(2097152);       // recon MLP hidden [16384,128]
    float* bPL  = alloc(16384);         // pool logits [8,2048]
    float* bPM  = alloc(512);           // pooled [8,64]
    float* bPH  = alloc(512);           // head hidden [8,64]

    // ---- helpers ----
    auto gemm = [&](long long M, long long N, long long K,
                    const float* A, long long lda, long long sA,
                    const float* B, long long ldb, long long sB, int transB,
                    float* C, long long ldc, long long sC,
                    float alpha, const float* bias, int act, int batch) {
        dim3 g((unsigned)((N + 63) / 64), (unsigned)((M + 31) / 32), (unsigned)batch);
        wmma_gemm_k<<<g, 128, 0, stream>>>((int)M, (int)N, (int)K,
                                           A, lda, sA, B, ldb, sB, transB,
                                           C, ldc, sC, alpha, bias, act);
    };
    auto conv = [&](const float* x, const float* w, const float* b, float* y,
                    int Cin, int Cout, int Lin, int Lout, int Kw, int strd, int pad, int act) {
        long long tot = 8LL * Cout * Lout;
        conv1d_k<<<(unsigned)((tot + 255) / 256), 256, 0, stream>>>(
            x, w, b, y, 8, Cin, Cout, Lin, Lout, Kw, strd, pad, act);
    };
    auto film = [&](float* x, const float* fw, const float* fb, int C, int L) {
        gemm(8, 2 * C, 8, bSE, 8, 0, fw, 2 * C, 0, 0, bGB, 2 * C, 0, 1.0f, fb, 0, 1);
        long long tot = 8LL * C * L;
        film_k<<<(unsigned)((tot + 255) / 256), 256, 0, stream>>>(x, bGB, C, L, 8);
    };
    auto ssm = [&](const float* x, float* y, const float* A, const float* Bm,
                   const float* Cm, const float* Dv, int C, int L) {
        long long tot = (long long)C * L;
        ssm_kfun_k<<<(unsigned)((tot + 255) / 256), 256, 0, stream>>>(A, Bm, Cm, bK, C, 64, L);
        dim3 g((unsigned)((L + 255) / 256), (unsigned)C, 8);
        ssm_apply_k<<<g, 256, 0, stream>>>(x, bK, Dv, y, C, L);
    };
    auto transpose = [&](const float* x, float* y, int C, int L, int dir) {
        long long tot = 8LL * C * L;
        transpose_k<<<(unsigned)((tot + 255) / 256), 256, 0, stream>>>(x, y, 8, C, L, dir);
    };
    // post-norm transformer block on t [8,L,D], in-place; pb = index of 'wq'
    auto transformer = [&](float* t, int L, int D, int pb) {
        const int H = 8, dh = D / 8;
        const long long BL = 8LL * L;
        gemm(BL, D, D, t, D, 0, in(pb + 0), D, 0, 0, bQ,  D, 0, 1.0f, nullptr, 0, 1);
        gemm(BL, D, D, t, D, 0, in(pb + 1), D, 0, 0, bKK, D, 0, 1.0f, nullptr, 0, 1);
        gemm(BL, D, D, t, D, 0, in(pb + 2), D, 0, 0, bV,  D, 0, 1.0f, nullptr, 0, 1);
        const float al = 1.0f / sqrtf((float)dh);
        for (int b = 0; b < 8; ++b) {
            long long bo = (long long)b * L * D;
            gemm(L, L, dh, bQ + bo, D, dh, bKK + bo, D, dh, 1,
                 bSC, L, (long long)L * L, al, nullptr, 0, H);
            softmax_rows_k<<<(unsigned)(H * L), 256, 0, stream>>>(bSC, L);
            gemm(L, dh, L, bSC, L, (long long)L * L, bV + bo, D, dh, 0,
                 bTMP + bo, D, dh, 1.0f, nullptr, 0, H);
        }
        gemm(BL, D, D, bTMP, D, 0, in(pb + 3), D, 0, 0, bQ, D, 0, 1.0f, nullptr, 0, 1);
        layernorm_k<<<(unsigned)BL, 256, 0, stream>>>(t, bQ, in(pb + 4), in(pb + 5), t, D);
        gemm(BL, 2 * D, D, t, D, 0, in(pb + 8), 2 * D, 0, 0, bFF, 2 * D, 0, 1.0f, in(pb + 9), 1, 1);
        gemm(BL, D, 2 * D, bFF, 2 * D, 0, in(pb + 10), D, 0, 0, bKK, D, 0, 1.0f, in(pb + 11), 0, 1);
        layernorm_k<<<(unsigned)BL, 256, 0, stream>>>(t, bKK, in(pb + 6), in(pb + 7), t, D);
    };

    // ---- se = gelu(static @ se_w + se_b) ----
    gemm(8, 8, 4, SP, 4, 0, in(2), 8, 0, 0, bSE, 8, 0, 1.0f, in(3), 1, 1);

    // ---- encoder: param base 4 + 10*i ----
    const int CHs[4]  = {64, 128, 256, 512};
    const int CINs[4] = {1, 64, 128, 256};
    const float* cur = X;
    for (int i = 0; i < 4; ++i) {
        int pb = 4 + i * 10;
        int c = CHs[i], cin = CINs[i], L = 2048 >> i;
        conv(cur, in(pb + 0), in(pb + 1), P0, cin, c, L, L, 3, 1, 1, 1);      // gelu conv
        film(P0, in(pb + 2), in(pb + 3), c, L);
        ssm(P0, SK[i], in(pb + 4), in(pb + 5), in(pb + 6), in(pb + 7), c, L); // h + ssm(h)
        conv(SK[i], in(pb + 8), in(pb + 9), P1, c, c, L, L / 2, 3, 2, 1, 0);  // stride-2 down
        cur = P1;
    }

    // ---- bottleneck: base 44 (wq..ff_b2=44..55, film 56/57, A..D 58..61) ----
    {
        int D = 512, L = 128;
        film(P1, in(56), in(57), D, L);
        transpose(P1, bT, D, L, 0);
        transformer(bT, L, D, 44);
        transpose(bT, P0, D, L, 1);
        ssm(P0, P1, in(58), in(59), in(60), in(61), D, L);
    }

    // ---- decoder: bases {62,68,86,104}; transformers at i>=1 (pb+6) ----
    const int decb[4] = {62, 68, 86, 104};
    float* cur2 = P1;
    for (int i = 3; i >= 0; --i) {
        int c = CHs[i];
        int cinD = CHs[i < 3 ? i + 1 : 3];
        int Lb = 2048 >> (i + 1), La = Lb * 2;
        int pb = decb[i];
        {
            long long tot = 8LL * cinD * La;
            upsample2_k<<<(unsigned)((tot + 255) / 256), 256, 0, stream>>>(cur2, P0, 8, cinD, Lb);
        }
        conv(P0, in(pb + 0), in(pb + 1), P1, cinD, c, La, La, 3, 1, 1, 1);    // up conv, gelu
        {
            long long tot = 8LL * 2 * c * La;
            concat_k<<<(unsigned)((tot + 255) / 256), 256, 0, stream>>>(P1, SK[i], P2, 8, c, c, La);
        }
        conv(P2, in(pb + 2), in(pb + 3), P0, 2 * c, c, La, La, 3, 1, 1, 1);   // fuse conv, gelu
        film(P0, in(pb + 4), in(pb + 5), c, La);
        if (i >= 1) {
            transpose(P0, bT, c, La, 0);
            transformer(bT, La, c, pb + 6);
            transpose(bT, P1, c, La, 1);
            cur2 = P1;
        } else {
            cur2 = P0;
        }
    }

    // ---- output projection conv k=7 (base 122) ----
    conv(cur2, in(122), in(123), P1, 64, 64, 2048, 2048, 7, 1, 3, 1);   // [8,64,2048]
    transpose(P1, bT, 64, 2048, 0);                                     // ht [8,2048,64]

    // recon = (gelu(ht@sig_w1+b1) @ sig_w2 + b2)[...,0]
    gemm(16384, 128, 64, bT, 64, 0, in(124), 128, 0, 0, bSIG, 128, 0, 1.0f, in(125), 1, 1);
    gemm(16384, 1, 128, bSIG, 128, 0, in(126), 1, 0, 0, OUT, 1, 0, 1.0f, in(127), 0, 1);

    auto pool = [&](const float* wv, float* outp) {
        pool_logits_k<<<(unsigned)((8 * 2048 + 255) / 256), 256, 0, stream>>>(P1, wv, bPL, 8, 64, 2048);
        softmax_rows_k<<<8, 256, 0, stream>>>(bPL, 2048);
        pool_reduce_k<<<2, 256, 0, stream>>>(P1, bPL, outp, 8, 64, 2048);
    };

    // peak heads
    pool(in(128), bPM);
    gemm(8, 64, 64, bPM, 64, 0, in(129), 64, 0, 0, bPH, 64, 0, 1.0f, in(130), 1, 1);
    gemm(8, 1, 64, bPH, 64, 0, in(131), 1, 0, 0, OUT + 16384, 1, 0, 1.0f, in(132), 0, 1);
    gemm(8, 1, 64, bPH, 64, 0, in(133), 1, 0, 0, OUT + 16392, 1, 0, 1.0f, in(134), 0, 1);
    gemm(8, 1, 64, bPH, 64, 0, in(135), 1, 0, 0, OUT + 16400, 1, 0, 1.0f, in(136), 0, 1);
    // class head
    pool(in(137), bPM);
    gemm(8, 64, 64, bPM, 64, 0, in(138), 64, 0, 0, bPH, 64, 0, 1.0f, in(139), 1, 1);
    gemm(8, 5, 64, bPH, 64, 0, in(140), 5, 0, 0, OUT + 16408, 5, 0, 1.0f, in(141), 0, 1);
    // threshold head (sigmoid * 120)
    pool(in(142), bPM);
    gemm(8, 64, 64, bPM, 64, 0, in(143), 64, 0, 0, bPH, 64, 0, 1.0f, in(144), 1, 1);
    gemm(8, 1, 64, bPH, 64, 0, in(145), 1, 0, 0, OUT + 16448, 1, 0, 1.0f, in(146), 2, 1);
}
